// DGCRN_63565515981216
// MI455X (gfx1250) — compile-verified, matrix operand order checked by reading, not run on hardware
//
#include <hip/hip_runtime.h>
#include <hip/hip_bf16.h>
#include <math.h>

// ---------------- static model dims (match reference) ----------------
#define BATCH     16
#define NNODES    400
#define HIDN      64
#define INDIM     2
#define CIN       66      // INDIM + HIDN
#define DINC      198     // 3 * CIN
#define NODE_DIM  40
#define HYPER_DIM 16
#define MID_DIM   2
#define SEQ       12
#define BN        (BATCH * NNODES)   // 6400

typedef __attribute__((ext_vector_type(16))) _Float16 v16h;
typedef __attribute__((ext_vector_type(8)))  _Float16 v8h;
typedef __attribute__((ext_vector_type(2)))  _Float16 v2h;
typedef __attribute__((ext_vector_type(8)))  float    v8f;

static inline int ceil_div(int a, int b) { return (a + b - 1) / b; }

__device__ __forceinline__ float sigmoidf_(float x) { return 1.0f / (1.0f + __expf(-x)); }

// =====================================================================
// LDS-staged WMMA GEMM:  C = alpha * op(A)*op(B) + beta * C
//   logical A: [M,K]  element (m,k) at A[ TA ? k*lda+m : m*lda+k ]
//   logical B: [K,N]  element (k,n) at B[ TB ? n*ldb+k : k*ldb+n ]
//   C row-major [M,N], ldc.  blockIdx.z = batch.
// Block = 256 threads = 8 waves; macro-tile 64(M) x 32(N); each wave owns
// one 16x16 tile.  K consumed 32/step through f16 LDS tiles:
//   Alds[64][40], Blds[32][40]  (K-contiguous rows, 80B stride: 16B-aligned
//   for ds_load_b128, bank-staggered).
// Interior tiles (block-uniform test) take an unguarded vectorized path:
//   float2 global loads -> f16 pair -> one ds_store_b32 per element-pair.
// Edge tiles use clamped addresses + arithmetic masking (v * mask) so the
// loads stay unconditional -- no EXEC-mask divergence.
// Fragment layout per CDNA5 ISA 7.12.2 (16-bit 16x32):
//   lane = half*16 + r15 ; elems 0..7 -> K=half*8+i ; 8..15 -> K=16+half*8+i
// =====================================================================
#define LDSA_STRIDE 40
#define LDSB_STRIDE 40

template <int TA, int TB>
__global__ __launch_bounds__(256) void wmma_gemm(
    const float* __restrict__ A, long long batchA, int lda,
    const float* __restrict__ B, long long batchB, int ldb,
    float* __restrict__ C, long long batchC, int ldc,
    int M, int N, int K, float alpha, float beta)
{
  __shared__ __attribute__((aligned(16))) _Float16 Alds[64 * LDSA_STRIDE];
  __shared__ __attribute__((aligned(16))) _Float16 Blds[32 * LDSB_STRIDE];

  const int bz = blockIdx.z;
  const float* Ab = A + (long long)bz * batchA;
  const float* Bb = B + (long long)bz * batchB;
  float*       Cb = C + (long long)bz * batchC;

  const int m0   = blockIdx.x * 64;
  const int n0   = blockIdx.y * 32;
  const int tid  = threadIdx.x;
  const int wave = tid >> 5;
  const int lane = tid & 31;
  const int half = lane >> 4;
  const int l15  = lane & 15;
  const int wm   = wave >> 1;   // 0..3 : M-subtile
  const int wn   = wave & 1;    // 0..1 : N-subtile

  const bool mFull = (m0 + 64 <= M);
  const bool nFull = (n0 + 32 <= N);

  v8f acc = {};

  for (int k0 = 0; k0 < K; k0 += 32) {
    const bool kFull = (k0 + 32 <= K);
    __syncthreads();

    // ---- cooperative load A tile: 64(M) x 32(K) ----
    if (mFull & kFull) {
      // 1024 f16-pairs, 4 per thread; float2 loads (8B-aligned: lda,k0 even)
#pragma unroll
      for (int r = 0; r < 4; ++r) {
        int p = tid + r * 256;
        int ml, kp;
        if (TA) { ml = p & 63; kp = (p >> 6) << 1; }   // phys [K,M]: m contiguous
        else    { ml = p >> 4; kp = (p & 15) << 1; }   // phys [M,K]: k contiguous
        int mm = m0 + ml, kk = k0 + kp;
        float x0, x1;
        if (TA) { x0 = Ab[kk * lda + mm]; x1 = Ab[(kk + 1) * lda + mm]; }
        else    { float2 t = *(const float2*)(Ab + mm * lda + kk); x0 = t.x; x1 = t.y; }
        v2h h; h[0] = (_Float16)x0; h[1] = (_Float16)x1;
        *(v2h*)(Alds + ml * LDSA_STRIDE + kp) = h;
      }
    } else {
#pragma unroll
      for (int r = 0; r < 8; ++r) {
        int flat = tid + r * 256;
        int ml, kl;
        if (TA) { ml = flat & 63; kl = flat >> 6; }
        else    { ml = flat >> 5; kl = flat & 31; }
        int mm = m0 + ml, kk = k0 + kl;
        int mc = mm < M ? mm : M - 1;
        int kc = kk < K ? kk : K - 1;
        int idx = TA ? kc * lda + mc : mc * lda + kc;
        float v = Ab[idx];
        float msk = (float)((mm < M) & (kk < K));
        Alds[ml * LDSA_STRIDE + kl] = (_Float16)(v * msk);
      }
    }

    // ---- cooperative load B tile: 32(K) x 32(N), stored as [n][k] ----
    if (nFull & kFull) {
#pragma unroll
      for (int r = 0; r < 2; ++r) {
        int p = tid + r * 256;
        int nl, kp;
        if (TB) { nl = p >> 4; kp = (p & 15) << 1; }   // phys [N,K]: k contiguous
        else    { nl = p & 31; kp = (p >> 5) << 1; }   // phys [K,N]: n contiguous
        int nn = n0 + nl, kk = k0 + kp;
        float x0, x1;
        if (TB) { float2 t = *(const float2*)(Bb + nn * ldb + kk); x0 = t.x; x1 = t.y; }
        else    { x0 = Bb[kk * ldb + nn]; x1 = Bb[(kk + 1) * ldb + nn]; }
        v2h h; h[0] = (_Float16)x0; h[1] = (_Float16)x1;
        *(v2h*)(Blds + nl * LDSB_STRIDE + kp) = h;
      }
    } else {
#pragma unroll
      for (int r = 0; r < 4; ++r) {
        int flat = tid + r * 256;
        int nl, kl;
        if (TB) { nl = flat >> 5; kl = flat & 31; }
        else    { nl = flat & 31; kl = flat >> 5; }
        int nn = n0 + nl, kk = k0 + kl;
        int nc = nn < N ? nn : N - 1;
        int kc = kk < K ? kk : K - 1;
        int idx = TB ? nc * ldb + kc : kc * ldb + nc;
        float v = Bb[idx];
        float msk = (float)((nn < N) & (kk < K));
        Blds[nl * LDSB_STRIDE + kl] = (_Float16)(v * msk);
      }
    }

    // prefetch next K-chunk (global_prefetch_b8; speculative)
    if (k0 + 32 < K) {
      int pa = TA ? (k0 + 32) * lda + (m0 + (tid & 63))
                  : (m0 + (tid >> 2)) * lda + (k0 + 32);
      int pb = TB ? (n0 + (tid & 31)) * ldb + (k0 + 32)
                  : (k0 + 32 + (tid >> 5)) * ldb + n0;
      __builtin_prefetch(Ab + pa, 0, 1);
      __builtin_prefetch(Bb + pb, 0, 1);
    }
    __syncthreads();

    // ---- fragment reads (2x ds_load_b128 each) + WMMA ----
    const _Float16* Ar = &Alds[(wm * 16 + l15) * LDSA_STRIDE];
    const _Float16* Br = &Blds[(wn * 16 + l15) * LDSB_STRIDE];
    v8h a_lo = *(const v8h*)(Ar + half * 8);
    v8h a_hi = *(const v8h*)(Ar + 16 + half * 8);
    v8h b_lo = *(const v8h*)(Br + half * 8);
    v8h b_hi = *(const v8h*)(Br + 16 + half * 8);
    v16h af, bf;
#pragma unroll
    for (int i = 0; i < 8; ++i) {
      af[i] = a_lo[i]; af[8 + i] = a_hi[i];
      bf[i] = b_lo[i]; bf[8 + i] = b_hi[i];
    }
    acc = __builtin_amdgcn_wmma_f32_16x16x32_f16(false, af, false, bf,
                                                 (short)0, acc, false, false);
  }

  // ---- store: lanes 0-15 -> M=v, lanes 16-31 -> M=v+8 ; N = lane&15 ----
  const int tm = m0 + wm * 16;
  const int tn = n0 + wn * 16;
#pragma unroll
  for (int v = 0; v < 8; ++v) {
    int mm = tm + v + 8 * half;
    int nn = tn + l15;
    if (mm < M && nn < N) {
      int idx = mm * ldc + nn;
      float prev = (beta != 0.f) ? Cb[idx] : 0.f;
      Cb[idx] = alpha * acc[v] + beta * prev;
    }
  }
}

// ============================ elementwise ============================
__global__ void k_fill(float* __restrict__ p, int n, float v)
{
  int i = blockIdx.x * blockDim.x + threadIdx.x;
  if (i < n) p[i] = v;
}

// hyper_in / comb = concat([x, H], -1) -> [B,N,66]
__global__ void k_concat_xH(const float* __restrict__ x, int xStrideB,
                            const float* __restrict__ H, float* __restrict__ out)
{
  int idx = blockIdx.x * blockDim.x + threadIdx.x;
  if (idx >= BN * CIN) return;
  int c = idx % CIN;
  int bn = idx / CIN;
  int n = bn % NNODES, b = bn / NNODES;
  out[idx] = (c < INDIM) ? x[b * xStrideB + n * INDIM + c]
                         : H[bn * HIDN + (c - INDIM)];
}

// ho[...,0:66]=src ; ho[...,66:132]=0.05*src ; ho[...,132:198]=0.05*src
__global__ void k_seed_ho(const float* __restrict__ src, float* __restrict__ ho)
{
  int idx = blockIdx.x * blockDim.x + threadIdx.x;
  if (idx >= BN * DINC) return;
  int c = idx % DINC;
  int bn = idx / DINC;
  float v = src[bn * CIN + (c % CIN)];
  ho[idx] = (c < CIN) ? v : 0.05f * v;
}

// sigmoid(fc1+b1) -> sigmoid(fc2) -> fc3 ; optional accumulate into f
__global__ void k_hyper_mlp(const float* __restrict__ fc1raw,
                            const float* __restrict__ b1,
                            const float* __restrict__ w2, const float* __restrict__ b2,
                            const float* __restrict__ w3, const float* __restrict__ b3,
                            float* __restrict__ f, int accumulate)
{
  int bn = blockIdx.x * blockDim.x + threadIdx.x;
  if (bn >= BN) return;
  float h1[HYPER_DIM];
#pragma unroll
  for (int i = 0; i < HYPER_DIM; ++i)
    h1[i] = sigmoidf_(fc1raw[bn * HYPER_DIM + i] + b1[i]);
  float h2[MID_DIM];
#pragma unroll
  for (int j = 0; j < MID_DIM; ++j) {
    float s = b2[j];
#pragma unroll
    for (int i = 0; i < HYPER_DIM; ++i) s += h1[i] * w2[i * MID_DIM + j];
    h2[j] = sigmoidf_(s);
  }
  for (int d = 0; d < NODE_DIM; ++d) {
    float o = b3[d] + h2[0] * w3[0 * NODE_DIM + d] + h2[1] * w3[1 * NODE_DIM + d];
    int i = bn * NODE_DIM + d;
    f[i] = accumulate ? (f[i] + o) : o;
  }
}

// nv = tanh(3 * emb * f)
__global__ void k_nodevec(const float* __restrict__ f1, const float* __restrict__ f2,
                          const float* __restrict__ emb1, const float* __restrict__ emb2,
                          float* __restrict__ nv1, float* __restrict__ nv2)
{
  int idx = blockIdx.x * blockDim.x + threadIdx.x;
  if (idx >= BN * NODE_DIM) return;
  int d = idx % NODE_DIM;
  int n = (idx / NODE_DIM) % NNODES;
  nv1[idx] = tanhf(3.0f * emb1[n * NODE_DIM + d] * f1[idx]);
  nv2[idx] = tanhf(3.0f * emb2[n * NODE_DIM + d] * f2[idx]);
}

// adj = relu(tanh(3*(G - G^T))) ; out row n of norm_adj(adj (+T)) per block
__global__ void k_norm_adj(const float* __restrict__ G, float* __restrict__ out,
                           int transposed)
{
  int n = blockIdx.x, b = blockIdx.y;
  const float* Gb = G + b * NNODES * NNODES;
  float* Ob = out + (b * NNODES + n) * NNODES;
  __shared__ float red[256];
  float s = 0.f;
  for (int m = threadIdx.x; m < NNODES; m += blockDim.x) {
    float g = transposed ? (Gb[m * NNODES + n] - Gb[n * NNODES + m])
                         : (Gb[n * NNODES + m] - Gb[m * NNODES + n]);
    float av = tanhf(3.0f * g); av = av > 0.f ? av : 0.f;
    s += av + (m == n ? 1.f : 0.f);
  }
  red[threadIdx.x] = s;
  __syncthreads();
  for (int off = 128; off > 0; off >>= 1) {
    if ((int)threadIdx.x < off) red[threadIdx.x] += red[threadIdx.x + off];
    __syncthreads();
  }
  float inv = 1.0f / red[0];
  for (int m = threadIdx.x; m < NNODES; m += blockDim.x) {
    float g = transposed ? (Gb[m * NNODES + n] - Gb[n * NNODES + m])
                         : (Gb[n * NNODES + m] - Gb[m * NNODES + n]);
    float av = tanhf(3.0f * g); av = av > 0.f ? av : 0.f;
    Ob[m] = (av + (m == n ? 1.f : 0.f)) * inv;
  }
}

// z = sigmoid(zraw+bz1+bz2) (in place), r likewise; temp = [x, r*H]
__global__ void k_gate_zr(float* __restrict__ zraw, float* __restrict__ rraw,
                          const float* __restrict__ bz1, const float* __restrict__ bz2,
                          const float* __restrict__ br1, const float* __restrict__ br2,
                          const float* __restrict__ x, int xStrideB,
                          const float* __restrict__ H, float* __restrict__ temp)
{
  int bn = blockIdx.x * blockDim.x + threadIdx.x;
  if (bn >= BN) return;
  int n = bn % NNODES, b = bn / NNODES;
  temp[bn * CIN + 0] = x[b * xStrideB + n * INDIM + 0];
  temp[bn * CIN + 1] = x[b * xStrideB + n * INDIM + 1];
  for (int h = 0; h < HIDN; ++h) {
    int i = bn * HIDN + h;
    float z = sigmoidf_(zraw[i] + bz1[h] + bz2[h]);
    float r = sigmoidf_(rraw[i] + br1[h] + br2[h]);
    zraw[i] = z;
    rraw[i] = r;
    temp[bn * CIN + INDIM + h] = r * H[i];
  }
}

// C = tanh(craw+bc1+bc2); H = z*H + (1-z)*C
__global__ void k_update_H(const float* __restrict__ craw,
                           const float* __restrict__ bc1, const float* __restrict__ bc2,
                           const float* __restrict__ z, float* __restrict__ H)
{
  int idx = blockIdx.x * blockDim.x + threadIdx.x;
  if (idx >= BN * HIDN) return;
  int h = idx % HIDN;
  float C = tanhf(craw[idx] + bc1[h] + bc2[h]);
  float zz = z[idx];
  H[idx] = zz * H[idx] + (1.0f - zz) * C;
}

// out[b,t,n,0] = H[b,n,:] @ w + bias ; prev = out
__global__ void k_head(const float* __restrict__ H, const float* __restrict__ wv,
                       const float* __restrict__ bias, float* __restrict__ out, int t,
                       float* __restrict__ prev)
{
  int bn = blockIdx.x * blockDim.x + threadIdx.x;
  if (bn >= BN) return;
  int n = bn % NNODES, b = bn / NNODES;
  float s = bias[0];
#pragma unroll
  for (int h = 0; h < HIDN; ++h) s += H[bn * HIDN + h] * wv[h];
  out[b * SEQ * NNODES + t * NNODES + n] = s;
  prev[bn] = s;
}

// decoder input: x = [prev, future[:,t,:,1]]
__global__ void k_dec_x(const float* __restrict__ prev, const float* __restrict__ future,
                        int t, float* __restrict__ x)
{
  int bn = blockIdx.x * blockDim.x + threadIdx.x;
  if (bn >= BN) return;
  int n = bn % NNODES, b = bn / NNODES;
  x[bn * INDIM + 0] = prev[bn];
  x[bn * INDIM + 1] = future[((b * SEQ + t) * NNODES + n) * INDIM + 1];
}

// ======================= host-side orchestration ======================
namespace {

struct Net {
  const float *hist, *future, *A1, *A2, *emb1, *emb2, *fw, *fb;
  const float* hyp[8][6];   // fc1w, fc1b, fc2w, fc2b, fc3w, fc3b
  const float* rnnW[12];
  const float* rnnB[12];
};

struct Ws {
  float *H, *hin, *temp, *ho, *fc1raw, *f1, *f2, *nv1, *nv2;
  float *zraw, *rraw, *craw, *G1, *adp, *adpT, *xdec, *prev;
};

void gemm(hipStream_t s, int ta, int tb,
          const float* A, long long bA, int lda,
          const float* B, long long bB, int ldb,
          float* C, long long bC, int ldc,
          int M, int N, int K, float alpha, float beta, int batch)
{
  dim3 grid(ceil_div(M, 64), ceil_div(N, 32), batch);
  dim3 blk(256, 1, 1);
  if (ta == 1 && tb == 0)
    wmma_gemm<1, 0><<<grid, blk, 0, s>>>(A, bA, lda, B, bB, ldb, C, bC, ldc,
                                         M, N, K, alpha, beta);
  else if (ta == 0 && tb == 1)
    wmma_gemm<0, 1><<<grid, blk, 0, s>>>(A, bA, lda, B, bB, ldb, C, bC, ldc,
                                         M, N, K, alpha, beta);
  else
    wmma_gemm<0, 0><<<grid, blk, 0, s>>>(A, bA, lda, B, bB, ldb, C, bC, ldc,
                                         M, N, K, alpha, beta);
}

// ho = concat([src, h1, h2]) with h_{d+1} = 0.05*src + 0.95*dyn^T h_d + 0.95*pre^T h_d
void rnn_prop(hipStream_t s, const float* src66, const float* adyn, const float* apre,
              float* ho)
{
  k_seed_ho<<<ceil_div(BN * DINC, 256), 256, 0, s>>>(src66, ho);
  for (int d = 0; d < 2; ++d) {
    const float* rd = ho + d * CIN;
    float*       wr = ho + (d + 1) * CIN;
    gemm(s, 1, 0, adyn, (long long)NNODES * NNODES, NNODES,
         rd, (long long)NNODES * DINC, DINC,
         wr, (long long)NNODES * DINC, DINC,
         NNODES, CIN, NNODES, 0.95f, 1.0f, BATCH);
    gemm(s, 1, 0, apre, 0, NNODES,
         rd, (long long)NNODES * DINC, DINC,
         wr, (long long)NNODES * DINC, DINC,
         NNODES, CIN, NNODES, 0.95f, 1.0f, BATCH);
  }
}

void step(hipStream_t s, const Net& net, const Ws& W, const float* xptr, int xStrideB,
          bool dec)
{
  const int ro  = dec ? 6 : 0;   // rnn param offset
  const int hyo = dec ? 4 : 0;   // hyper param offset

  // hyper_in = comb = [x, H]
  k_concat_xH<<<ceil_div(BN * CIN, 256), 256, 0, s>>>(xptr, xStrideB, W.H, W.hin);

  // ---- hypernetwork: f1, f2 (shared propagation per adjacency) ----
  for (int g = 0; g < 2; ++g) {
    const float* A = g ? net.A2 : net.A1;
    k_seed_ho<<<ceil_div(BN * DINC, 256), 256, 0, s>>>(W.hin, W.ho);
    for (int d = 0; d < 2; ++d) {
      gemm(s, 1, 0, A, 0, NNODES,
           W.ho + d * CIN, (long long)NNODES * DINC, DINC,
           W.ho + (d + 1) * CIN, (long long)NNODES * DINC, DINC,
           NNODES, CIN, NNODES, 0.95f, 1.0f, BATCH);
    }
    for (int j = 0; j < 2; ++j) {  // j=0 -> f1 (GCN1), j=1 -> f2 (GCN2)
      const float* const* hp = net.hyp[hyo + g * 2 + j];
      gemm(s, 0, 0, W.ho, 0, DINC, hp[0], 0, HYPER_DIM,
           W.fc1raw, 0, HYPER_DIM, BN, HYPER_DIM, DINC, 1.0f, 0.0f, 1);
      k_hyper_mlp<<<ceil_div(BN, 256), 256, 0, s>>>(
          W.fc1raw, hp[1], hp[2], hp[3], hp[4], hp[5], j ? W.f2 : W.f1, g);
    }
  }

  // ---- dynamic adjacency: G = nv1 @ nv2^T ; a = G - G^T (antisymmetric) ----
  k_nodevec<<<ceil_div(BN * NODE_DIM, 256), 256, 0, s>>>(W.f1, W.f2, net.emb1, net.emb2,
                                                         W.nv1, W.nv2);
  gemm(s, 0, 1, W.nv1, (long long)NNODES * NODE_DIM, NODE_DIM,
       W.nv2, (long long)NNODES * NODE_DIM, NODE_DIM,
       W.G1, (long long)NNODES * NNODES, NNODES,
       NNODES, NNODES, NODE_DIM, 1.0f, 0.0f, BATCH);
  k_norm_adj<<<dim3(NNODES, BATCH), 256, 0, s>>>(W.G1, W.adp, 0);
  k_norm_adj<<<dim3(NNODES, BATCH), 256, 0, s>>>(W.G1, W.adpT, 1);

  // ---- GRU gates z, r (gz/gr share propagated ho) ----
  rnn_prop(s, W.hin, W.adp, net.A1, W.ho);
  gemm(s, 0, 0, W.ho, 0, DINC, net.rnnW[ro + 0], 0, HIDN, W.zraw, 0, HIDN,
       BN, HIDN, DINC, 1.0f, 0.0f, 1);
  gemm(s, 0, 0, W.ho, 0, DINC, net.rnnW[ro + 2], 0, HIDN, W.rraw, 0, HIDN,
       BN, HIDN, DINC, 1.0f, 0.0f, 1);
  rnn_prop(s, W.hin, W.adpT, net.A2, W.ho);
  gemm(s, 0, 0, W.ho, 0, DINC, net.rnnW[ro + 1], 0, HIDN, W.zraw, 0, HIDN,
       BN, HIDN, DINC, 1.0f, 1.0f, 1);
  gemm(s, 0, 0, W.ho, 0, DINC, net.rnnW[ro + 3], 0, HIDN, W.rraw, 0, HIDN,
       BN, HIDN, DINC, 1.0f, 1.0f, 1);
  k_gate_zr<<<ceil_div(BN, 256), 256, 0, s>>>(
      W.zraw, W.rraw, net.rnnB[ro + 0], net.rnnB[ro + 1], net.rnnB[ro + 2],
      net.rnnB[ro + 3], xptr, xStrideB, W.H, W.temp);

  // ---- candidate C and hidden update ----
  rnn_prop(s, W.temp, W.adp, net.A1, W.ho);
  gemm(s, 0, 0, W.ho, 0, DINC, net.rnnW[ro + 4], 0, HIDN, W.craw, 0, HIDN,
       BN, HIDN, DINC, 1.0f, 0.0f, 1);
  rnn_prop(s, W.temp, W.adpT, net.A2, W.ho);
  gemm(s, 0, 0, W.ho, 0, DINC, net.rnnW[ro + 5], 0, HIDN, W.craw, 0, HIDN,
       BN, HIDN, DINC, 1.0f, 1.0f, 1);
  k_update_H<<<ceil_div(BN * HIDN, 256), 256, 0, s>>>(
      W.craw, net.rnnB[ro + 4], net.rnnB[ro + 5], W.zraw, W.H);
}

}  // namespace

extern "C" void kernel_launch(void* const* d_in, const int* in_sizes, int n_in,
                              void* d_out, int out_size, void* d_ws, size_t ws_size,
                              hipStream_t stream)
{
  (void)in_sizes; (void)n_in; (void)out_size; (void)ws_size;

  // ---- input mapping: setup_inputs() insertion order, params flattened in
  // insertion order ('w' before 'b' as _lin builds them):
  //  0 history_data  1 future_data  2 A1  3 A2
  //  4 emb1  5 emb2  6 fc_final.w  7 fc_final.b
  //  8.. hyper blocks x8 (GCN1_tg, GCN2_tg, GCN1_tg_1, GCN2_tg_1,
  //                       GCN1_tg_de, GCN2_tg_de, GCN1_tg_de_1, GCN2_tg_de_1)
  //       each: fc1.w fc1.b fc2.w fc2.b fc3.w fc3.b
  //  56.. rnn x12 (gz1,gz2,gr1,gr2,gc1,gc2, then _de) each: w, b
  //  80 task_level (== 12, hardcoded)
  Net net;
  net.hist   = (const float*)d_in[0];
  net.future = (const float*)d_in[1];
  net.A1     = (const float*)d_in[2];
  net.A2     = (const float*)d_in[3];
  net.emb1   = (const float*)d_in[4];
  net.emb2   = (const float*)d_in[5];
  net.fw     = (const float*)d_in[6];
  net.fb     = (const float*)d_in[7];
  for (int h = 0; h < 8; ++h)
    for (int k = 0; k < 6; ++k) net.hyp[h][k] = (const float*)d_in[8 + h * 6 + k];
  for (int i = 0; i < 12; ++i) {
    net.rnnW[i] = (const float*)d_in[56 + i * 2];
    net.rnnB[i] = (const float*)d_in[57 + i * 2];
  }

  // ---- workspace layout (floats), ~50.3 MB total ----
  float* p = (float*)d_ws;
  Ws W;
  W.H      = p; p += BN * HIDN;            // 409600
  W.hin    = p; p += BN * CIN;             // 422400
  W.temp   = p; p += BN * CIN;             // 422400
  W.ho     = p; p += BN * DINC;            // 1267200
  W.fc1raw = p; p += BN * HYPER_DIM;       // 102400
  W.f1     = p; p += BN * NODE_DIM;        // 256000
  W.f2     = p; p += BN * NODE_DIM;
  W.nv1    = p; p += BN * NODE_DIM;
  W.nv2    = p; p += BN * NODE_DIM;
  W.zraw   = p; p += BN * HIDN;
  W.rraw   = p; p += BN * HIDN;
  W.craw   = p; p += BN * HIDN;
  W.G1     = p; p += (long long)BATCH * NNODES * NNODES;   // 2.56M each
  W.adp    = p; p += (long long)BATCH * NNODES * NNODES;
  W.adpT   = p; p += (long long)BATCH * NNODES * NNODES;
  W.xdec   = p; p += BN * INDIM;
  W.prev   = p; p += BN;

  float* out = (float*)d_out;

  k_fill<<<ceil_div(BN * HIDN, 256), 256, 0, stream>>>(W.H, BN * HIDN, 0.f);
  k_fill<<<ceil_div(BN, 256), 256, 0, stream>>>(W.prev, BN, 0.f);

  // encoder: x_t = history_data[:, t]  ([B,T,N,2] -> base + t*N*2, strideB = T*N*2)
  for (int t = 0; t < SEQ; ++t)
    step(stream, net, W, net.hist + t * NNODES * INDIM,
         SEQ * NNODES * INDIM, false);

  // decoder
  for (int t = 0; t < SEQ; ++t) {
    k_dec_x<<<ceil_div(BN, 256), 256, 0, stream>>>(W.prev, net.future, t, W.xdec);
    step(stream, net, W, W.xdec, NNODES * INDIM, true);
    k_head<<<ceil_div(BN, 256), 256, 0, stream>>>(W.H, net.fw, net.fb, out, t, W.prev);
  }
}
